// StandardAttention_62173946576909
// MI455X (gfx1250) — 16-bit and 8-bit matrix operand order repaired and checked by reading, compiled, not run on hardware
//
#include <hip/hip_runtime.h>
#include <cstdint>


#define DEV static __device__ __forceinline__

typedef __attribute__((ext_vector_type(8)))  int    v8i;
typedef __attribute__((ext_vector_type(8)))  float  v8f;
typedef __attribute__((ext_vector_type(16))) __bf16 v16bf;

union I8Frag  { v8i   v; unsigned u[8]; };
union BF16Frag{ v16bf v; unsigned u[8]; };

DEV unsigned short f2bf(float f) {
    unsigned u = __float_as_uint(f);
    u += 0x7FFFu + ((u >> 16) & 1u);
    return (unsigned short)(u >> 16);
}

DEV v8f wmma_bf16(const v16bf& a, const v16bf& b, v8f c) {
    return __builtin_amdgcn_wmma_f32_16x16x32_bf16(false, a, false, b,
                                                   (short)0, c, false, false);
}

DEV v8i wmma_iu8(const v8i& a, const v8i& b, v8i c) {
    return __builtin_amdgcn_wmma_i32_16x16x64_iu8(true, a, true, b, c,
                                                  false, false);
}

DEV void async_load_b128(unsigned lds_addr, const void* gaddr) {
    asm volatile("global_load_async_to_lds_b128 %0, %1, off"
                 :: "v"(lds_addr), "v"((unsigned long long)(size_t)gaddr)
                 : "memory");
}
DEV void wait_async0() {
    asm volatile("s_wait_asynccnt 0x0" ::: "memory");
}

DEV unsigned lds_addr_of(const void* p) {
    return (unsigned)(size_t)p;
}

__global__ void batch_stats_kernel(const float* __restrict__ x, float* out, long n)
{
    const long b = blockIdx.y;
    const float* p = x + b * n;
    float s = 0.f, s2 = 0.f;
    for (long i = (long)blockIdx.x * blockDim.x + threadIdx.x; i < n;
         i += (long)gridDim.x * blockDim.x) {
        float v = p[i]; s += v; s2 += v * v;
    }
    __shared__ float rs[256], rs2[256];
    rs[threadIdx.x] = s; rs2[threadIdx.x] = s2; __syncthreads();
    for (int o = 128; o > 0; o >>= 1) {
        if ((int)threadIdx.x < o) {
            rs[threadIdx.x]  += rs[threadIdx.x + o];
            rs2[threadIdx.x] += rs2[threadIdx.x + o];
        }
        __syncthreads();
    }
    if (threadIdx.x == 0) {
        atomicAdd(&out[b * 2 + 0], rs[0]);
        atomicAdd(&out[b * 2 + 1], rs2[0]);
    }
}

__global__ void wabs_kernel(const float* w0, const float* w1,
                            const float* w2, const float* w3, float* out)
{
    const float* ws[4] = {w0, w1, w2, w3};
    const int wi = blockIdx.y;
    const float* p = ws[wi];
    const long n = 1024L * 1024L;
    float s = 0.f;
    for (long i = (long)blockIdx.x * blockDim.x + threadIdx.x; i < n;
         i += (long)gridDim.x * blockDim.x)
        s += fabsf(p[i]);
    __shared__ float rs[256];
    rs[threadIdx.x] = s; __syncthreads();
    for (int o = 128; o > 0; o >>= 1) {
        if ((int)threadIdx.x < o) rs[threadIdx.x] += rs[threadIdx.x + o];
        __syncthreads();
    }
    if (threadIdx.x == 0) atomicAdd(&out[wi], rs[0]);
}

__global__ void wquant_kernel(const float* w0, const float* w1,
                              const float* w2, const float* w3,
                              const float* __restrict__ wsum,
                              int8_t* __restrict__ wq_out)
{
    const float* ws[4] = {w0, w1, w2, w3};
    const int wi = blockIdx.y;
    const float wmean = fmaxf(wsum[wi] * (1.0f / 1048576.0f), 1e-5f);
    const float inv = 1.0f / wmean;
    const long i = (long)blockIdx.x * blockDim.x + threadIdx.x;
    float q = rintf(ws[wi][i] * inv);
    q = fminf(fmaxf(q, -1.f), 1.f);
    wq_out[(long)wi * 1048576L + i] = (int8_t)q;
}

__global__ void act_quant_kernel(const float* __restrict__ x,
                                 const float* __restrict__ stats,
                                 int8_t* __restrict__ xq,
                                 float* __restrict__ ascale)
{
    const int row = blockIdx.x;
    const int b = row >> 11;
    const float inv_n = 1.0f / (2048.0f * 1024.0f);
    const float mu  = stats[b * 2 + 0] * inv_n;
    const float var = stats[b * 2 + 1] * inv_n - mu * mu;
    const float rstd = rsqrtf(var + 1e-5f);
    const float* xr = x + (long)row * 1024;

    float vals[4]; float am = 0.f;
    #pragma unroll
    for (int i = 0; i < 4; i++) {
        float v = (xr[threadIdx.x + 256 * i] - mu) * rstd;
        vals[i] = v; am = fmaxf(am, fabsf(v));
    }
    __shared__ float red[256];
    red[threadIdx.x] = am; __syncthreads();
    for (int o = 128; o > 0; o >>= 1) {
        if ((int)threadIdx.x < o)
            red[threadIdx.x] = fmaxf(red[threadIdx.x], red[threadIdx.x + o]);
        __syncthreads();
    }
    const float scale = 127.0f / fmaxf(red[0], 1e-5f);
    #pragma unroll
    for (int i = 0; i < 4; i++) {
        float q = rintf(vals[i] * scale);
        q = fminf(fmaxf(q, -128.f), 127.f);
        xq[(long)row * 1024 + threadIdx.x + 256 * i] = (int8_t)q;
    }
    if (threadIdx.x == 0) ascale[row] = scale;
}

__global__ void __launch_bounds__(256)
bitlinear_gemm_kernel(const int8_t* __restrict__ Aq,
                      const int8_t* __restrict__ Wq,
                      const float* __restrict__ ascale,
                      const float* __restrict__ wsum,
                      unsigned short* __restrict__ out_bf16,
                      float* __restrict__ out_f32, int mode)
{
    __shared__ int8_t Wlds[64 * 64];

    const int lane = threadIdx.x & 31;
    const int wave = threadIdx.x >> 5;
    const int hl   = lane >> 4;
    const int l15  = lane & 15;
    const int n0 = blockIdx.x * 64;
    const int m0 = blockIdx.y * 128 + wave * 16;

    const int wrow = threadIdx.x >> 2;
    const int wseg = (threadIdx.x & 3) * 16;
    const int8_t* wsrc = Wq + (long)(n0 + wrow) * 1024 + wseg;
    const unsigned wdst = lds_addr_of(Wlds) + wrow * 64 + wseg;

    const v8i z8 = {0,0,0,0,0,0,0,0};
    v8i acc0 = z8, acc1 = z8, acc2 = z8, acc3 = z8;

    const int8_t* arow = Aq + (long)(m0 + l15) * 1024;
    const int8_t* wl0 = Wlds + (l15)      * 64;
    const int8_t* wl1 = Wlds + (16 + l15) * 64;
    const int8_t* wl2 = Wlds + (32 + l15) * 64;
    const int8_t* wl3 = Wlds + (48 + l15) * 64;
    const int bko = hl * 16;

    #pragma unroll 1
    for (int kb = 0; kb < 1024; kb += 64) {
        async_load_b128(wdst, wsrc + kb);
        wait_async0();
        __syncthreads();

        I8Frag a;
        #pragma unroll
        for (int j = 0; j < 8; j++) {
            int k = kb + (j >> 2) * 32 + hl * 16 + (j & 3) * 4;
            a.u[j] = *(const unsigned*)(arow + k);
        }
        if (kb + 64 < 1024)
            __builtin_prefetch(arow + kb + 64, 0, 3);

        I8Frag b0, b1, b2, b3;
        #pragma unroll
        for (int j = 0; j < 8; j++) {
            int k = (j >> 2) * 32 + bko + (j & 3) * 4;
            b0.u[j] = *(const unsigned*)(wl0 + k);
            b1.u[j] = *(const unsigned*)(wl1 + k);
            b2.u[j] = *(const unsigned*)(wl2 + k);
            b3.u[j] = *(const unsigned*)(wl3 + k);
        }
        acc0 = wmma_iu8(a.v, b0.v, acc0);
        acc1 = wmma_iu8(a.v, b1.v, acc1);
        acc2 = wmma_iu8(a.v, b2.v, acc2);
        acc3 = wmma_iu8(a.v, b3.v, acc3);
        __syncthreads();
    }

    const float wmean = fmaxf(wsum[0] * (1.0f / 1048576.0f), 1e-5f);
    v8i accs[4] = {acc0, acc1, acc2, acc3};
    #pragma unroll
    for (int nt = 0; nt < 4; nt++) {
        #pragma unroll
        for (int r = 0; r < 8; r++) {
            const int m = m0 + r + hl * 8;
            const int n = n0 + nt * 16 + l15;
            const float y = (float)accs[nt][r] * wmean / ascale[m];
            if (mode == 0) {
                const int b = m >> 11, s = m & 2047, h = n >> 6, d = n & 63;
                out_bf16[((long)(b * 16 + h) * 2048 + s) * 64 + d] = f2bf(y);
            } else {
                out_f32[(long)m * 1024 + n] = y;
            }
        }
    }
}

__global__ void __launch_bounds__(256)
attn_kernel(const unsigned short* __restrict__ Qb,
            const unsigned short* __restrict__ Kb,
            const unsigned short* __restrict__ Vb,
            float* __restrict__ attn_out)
{
    __shared__ unsigned short Ktile[32 * 64];
    __shared__ unsigned short Vt[64 * 32];
    __shared__ unsigned short Plds[8 * 16 * 32];

    const int lane = threadIdx.x & 31;
    const int wave = threadIdx.x >> 5;
    const int hl   = lane >> 4;
    const int l15  = lane & 15;
    const int h = blockIdx.y;
    const int b = blockIdx.z;
    const int q0 = blockIdx.x * 128 + wave * 16;
    const long head_base = (long)(b * 16 + h) * 2048 * 64;

    const unsigned kdst = lds_addr_of(Ktile) + threadIdx.x * 16;

    BF16Frag qf[2];
    #pragma unroll
    for (int c = 0; c < 2; c++)
        #pragma unroll
        for (int j = 0; j < 8; j++) {
            int kelem = c * 32 + (j >> 2) * 16 + hl * 8 + (j & 3) * 2;
            qf[c].u[j] = *(const unsigned*)(Qb + head_base +
                                            (long)(q0 + l15) * 64 + kelem);
        }

    const v8f zf = {0.f,0.f,0.f,0.f,0.f,0.f,0.f,0.f};
    v8f o0 = zf, o1 = zf, o2 = zf, o3 = zf;
    float mi[8], li[8];
    #pragma unroll
    for (int r = 0; r < 8; r++) { mi[r] = -3.0e38f; li[r] = 0.f; }

    unsigned short* Pw = Plds + wave * 512;

    #pragma unroll 1
    for (int kt = 0; kt < 2048; kt += 32) {
        async_load_b128(kdst, Kb + head_base + (long)kt * 64 + threadIdx.x * 8);
        #pragma unroll
        for (int i = 0; i < 8; i++) {
            int idx = threadIdx.x * 8 + i;
            int key = idx >> 6, d = idx & 63;
            Vt[d * 32 + key] = Vb[head_base + (long)(kt + key) * 64 + d];
        }
        wait_async0();
        __syncthreads();

        v8f c0 = zf, c1 = zf;
        #pragma unroll
        for (int cc = 0; cc < 2; cc++) {
            BF16Frag kf0, kf1;
            #pragma unroll
            for (int j = 0; j < 8; j++) {
                int d = cc * 32 + (j >> 2) * 16 + hl * 8 + (j & 3) * 2;
                kf0.u[j] = *(const unsigned*)(Ktile + (l15)       * 64 + d);
                kf1.u[j] = *(const unsigned*)(Ktile + (16 + l15)  * 64 + d);
            }
            c0 = wmma_bf16(qf[cc].v, kf0.v, c0);
            c1 = wmma_bf16(qf[cc].v, kf1.v, c1);
        }

        #pragma unroll
        for (int r = 0; r < 8; r++) {
            float s0 = c0[r] * 0.125f;
            float s1 = c1[r] * 0.125f;
            float mv = fmaxf(s0, s1);
            #pragma unroll
            for (int off = 8; off > 0; off >>= 1)
                mv = fmaxf(mv, __shfl_xor(mv, off));
            float mnew  = fmaxf(mi[r], mv);
            float alpha = __expf(mi[r] - mnew);
            float p0 = __expf(s0 - mnew);
            float p1 = __expf(s1 - mnew);
            float ps = p0 + p1;
            #pragma unroll
            for (int off = 8; off > 0; off >>= 1)
                ps += __shfl_xor(ps, off);
            li[r] = li[r] * alpha + ps;
            mi[r] = mnew;
            o0[r] *= alpha; o1[r] *= alpha; o2[r] *= alpha; o3[r] *= alpha;
            const int prow = r + hl * 8;
            Pw[prow * 32 + l15]      = f2bf(p0);
            Pw[prow * 32 + 16 + l15] = f2bf(p1);
        }

        BF16Frag pf;
        #pragma unroll
        for (int j = 0; j < 8; j++) {
            int key = (j >> 2) * 16 + hl * 8 + (j & 3) * 2;
            pf.u[j] = *(const unsigned*)(Pw + l15 * 32 + key);
        }
        BF16Frag vf0, vf1, vf2, vf3;
        #pragma unroll
        for (int j = 0; j < 8; j++) {
            int key = (j >> 2) * 16 + hl * 8 + (j & 3) * 2;
            vf0.u[j] = *(const unsigned*)(Vt + (l15)      * 32 + key);
            vf1.u[j] = *(const unsigned*)(Vt + (16 + l15) * 32 + key);
            vf2.u[j] = *(const unsigned*)(Vt + (32 + l15) * 32 + key);
            vf3.u[j] = *(const unsigned*)(Vt + (48 + l15) * 32 + key);
        }
        o0 = wmma_bf16(pf.v, vf0.v, o0);
        o1 = wmma_bf16(pf.v, vf1.v, o1);
        o2 = wmma_bf16(pf.v, vf2.v, o2);
        o3 = wmma_bf16(pf.v, vf3.v, o3);
        __syncthreads();
    }

    v8f os[4] = {o0, o1, o2, o3};
    #pragma unroll
    for (int nt = 0; nt < 4; nt++)
        #pragma unroll
        for (int r = 0; r < 8; r++) {
            const int qq = q0 + r + hl * 8;
            attn_out[(long)(b * 2048 + qq) * 1024 + h * 64 + nt * 16 + l15] =
                os[nt][r] / li[r];
        }
}

extern "C" void kernel_launch(void* const* d_in, const int* in_sizes, int n_in,
                              void* d_out, int out_size, void* d_ws, size_t ws_size,
                              hipStream_t stream)
{
    const float* x   = (const float*)d_in[0];
    const float* w_q = (const float*)d_in[1];
    const float* w_k = (const float*)d_in[2];
    const float* w_v = (const float*)d_in[3];
    const float* w_o = (const float*)d_in[4];

    char* ws = (char*)d_ws;
    float*  stats    = (float*)ws;
    float*  ascale_x = (float*)(ws + 256);
    float*  ascale_a = (float*)(ws + 16640);
    int8_t* xq  = (int8_t*)(ws + 33024);
    int8_t* aq  = (int8_t*)(ws + 4227328);
    int8_t* wq8 = (int8_t*)(ws + 8421632);
    unsigned short* Qb = (unsigned short*)(ws + 12615936);
    unsigned short* Kb = (unsigned short*)(ws + 21004544);
    unsigned short* Vb = (unsigned short*)(ws + 29393152);
    float* attn = (float*)(ws + 37781760);

    hipMemsetAsync(d_ws, 0, 256, stream);

    batch_stats_kernel<<<dim3(64, 2), 256, 0, stream>>>(x, stats, 2048L * 1024L);
    wabs_kernel<<<dim3(64, 4), 256, 0, stream>>>(w_q, w_k, w_v, w_o, stats + 8);
    wquant_kernel<<<dim3(4096, 4), 256, 0, stream>>>(w_q, w_k, w_v, w_o, stats + 8, wq8);
    act_quant_kernel<<<4096, 256, 0, stream>>>(x, stats, xq, ascale_x);

    bitlinear_gemm_kernel<<<dim3(16, 32), 256, 0, stream>>>(
        xq, wq8 + 0L,        ascale_x, stats + 8,  Qb, nullptr, 0);
    bitlinear_gemm_kernel<<<dim3(16, 32), 256, 0, stream>>>(
        xq, wq8 + 1048576L,  ascale_x, stats + 9,  Kb, nullptr, 0);
    bitlinear_gemm_kernel<<<dim3(16, 32), 256, 0, stream>>>(
        xq, wq8 + 2097152L,  ascale_x, stats + 10, Vb, nullptr, 0);

    attn_kernel<<<dim3(16, 16, 2), 256, 0, stream>>>(Qb, Kb, Vb, attn);

    batch_stats_kernel<<<dim3(64, 2), 256, 0, stream>>>(attn, stats + 4, 2048L * 1024L);
    act_quant_kernel<<<4096, 256, 0, stream>>>(attn, stats + 4, aq, ascale_a);

    bitlinear_gemm_kernel<<<dim3(16, 32), 256, 0, stream>>>(
        aq, wq8 + 3145728L, ascale_a, stats + 11, nullptr, (float*)d_out, 1);
}
